// PointNet2Cls_84670985273358
// MI455X (gfx1250) — matrix operand order repaired and checked by reading, not compiled, not run on hardware
//
#include <hip/hip_runtime.h>
#include <hip/hip_bf16.h>
#include <math.h>

typedef __attribute__((ext_vector_type(16))) _Float16 v16h;
typedef __attribute__((ext_vector_type(8)))  _Float16 v8h;
typedef __attribute__((ext_vector_type(8)))  float    v8f;

#define BN_SCALE_C 0.9999950000374997f

// ---------------------------------------------------------------------------
// points [6,N] -> xyz [N,3] f32 (point-major), feat [N,3] f16
// ---------------------------------------------------------------------------
__global__ void split_points_kernel(const float* __restrict__ pts, int N,
                                    float* __restrict__ xyz, _Float16* __restrict__ feat) {
  int n = blockIdx.x * blockDim.x + threadIdx.x;
  if (n >= N) return;
  xyz[n * 3 + 0] = pts[0 * N + n];
  xyz[n * 3 + 1] = pts[1 * N + n];
  xyz[n * 3 + 2] = pts[2 * N + n];
  feat[n * 3 + 0] = (_Float16)pts[3 * N + n];
  feat[n * 3 + 1] = (_Float16)pts[4 * N + n];
  feat[n * 3 + 2] = (_Float16)pts[5 * N + n];
}

// f32 weights [Cout,Cin] -> f16 [Cout,Kpad] zero-padded (Kpad % 32 == 0)
__global__ void convert_weights_kernel(const float* __restrict__ W, _Float16* __restrict__ Wp,
                                       int Cout, int Cin, int Kpad) {
  int i = blockIdx.x * blockDim.x + threadIdx.x;
  if (i >= Cout * Kpad) return;
  int r = i / Kpad, c = i - r * Kpad;
  Wp[i] = (c < Cin) ? (_Float16)W[r * Cin + c] : (_Float16)0.0f;
}

// ---------------------------------------------------------------------------
// Farthest point sampling: single block, iterative; writes centroid coords.
// ---------------------------------------------------------------------------
__global__ __launch_bounds__(1024) void fps_kernel(const float* __restrict__ xyz, int N,
                                                   int npoint, float* __restrict__ ctrXyz,
                                                   float* __restrict__ minDist) {
  __shared__ float sd[1024];
  __shared__ int   si[1024];
  __shared__ float c3[3];
  __shared__ int   sCur;
  int tid = threadIdx.x;
  for (int j = tid; j < N; j += 1024) minDist[j] = 1e10f;
  if (tid == 0) sCur = 0;
  __syncthreads();
  for (int it = 0; it < npoint; ++it) {
    if (tid == 0) {
      int cur = sCur;
      c3[0] = xyz[cur * 3 + 0];
      c3[1] = xyz[cur * 3 + 1];
      c3[2] = xyz[cur * 3 + 2];
      ctrXyz[it * 3 + 0] = c3[0];
      ctrXyz[it * 3 + 1] = c3[1];
      ctrXyz[it * 3 + 2] = c3[2];
    }
    __syncthreads();
    float cx = c3[0], cy = c3[1], cz = c3[2];
    float bd = -1.0f; int bi = 0;
    for (int j = tid; j < N; j += 1024) {
      float dx = xyz[j * 3 + 0] - cx;
      float dy = xyz[j * 3 + 1] - cy;
      float dz = xyz[j * 3 + 2] - cz;
      float d = dx * dx + dy * dy + dz * dz;
      float md = fminf(minDist[j], d);
      minDist[j] = md;
      if (md > bd) { bd = md; bi = j; }
    }
    sd[tid] = bd; si[tid] = bi;
    __syncthreads();
    for (int s = 512; s > 0; s >>= 1) {
      if (tid < s && sd[tid + s] > sd[tid]) { sd[tid] = sd[tid + s]; si[tid] = si[tid + s]; }
      __syncthreads();
    }
    if (tid == 0) sCur = si[0];
    __syncthreads();
  }
}

// ---------------------------------------------------------------------------
// Ball query + grouping: one wave per centroid; ballot-based first-64 select,
// then write grouped (centered xyz ++ features, zero-padded to Kpad) as f16.
// ---------------------------------------------------------------------------
__global__ __launch_bounds__(256) void ball_group_kernel(
    const float* __restrict__ dXyz, int Nd,
    const _Float16* __restrict__ dFeat, int Cf,
    const float* __restrict__ ctr, int npt,
    float radius, _Float16* __restrict__ out, int Kpad) {
  __shared__ int sIdx[8][64];
  int lane = threadIdx.x & 31;
  int wv = threadIdx.x >> 5;
  int s = blockIdx.x * 8 + wv;
  if (s >= npt) return;
  float cx = ctr[s * 3 + 0], cy = ctr[s * 3 + 1], cz = ctr[s * 3 + 2];
  float r2 = radius * radius;
  int filled = 0;
  for (int base = 0; base < Nd && filled < 64; base += 32) {
    int j = base + lane;
    bool in = false;
    if (j < Nd) {
      float dx = dXyz[j * 3 + 0] - cx;
      float dy = dXyz[j * 3 + 1] - cy;
      float dz = dXyz[j * 3 + 2] - cz;
      in = (dx * dx + dy * dy + dz * dz) <= r2;
    }
    unsigned long long m = __ballot(in) & 0xffffffffull;
    int pre = __popcll(m & ((1ull << lane) - 1ull));
    if (in && (filled + pre) < 64) sIdx[wv][filled + pre] = j;
    filled += __popcll(m);
  }
  if (filled <= 0) { if (lane == 0) sIdx[wv][0] = 0; filled = 1; }
  if (filled > 64) filled = 64;
  for (int k = filled + lane; k < 64; k += 32) sIdx[wv][k] = sIdx[wv][0];
  int Cin = 3 + Cf;
  for (int t = lane; t < 64 * Kpad; t += 32) {
    int k = t / Kpad, c = t - k * Kpad;
    int j = sIdx[wv][k];
    float v = 0.0f;
    if (c < 3) {
      float ctrc = (c == 0) ? cx : ((c == 1) ? cy : cz);
      v = dXyz[j * 3 + c] - ctrc;
    } else if (c < Cin) {
      v = (float)dFeat[(size_t)j * Cf + (c - 3)];
    }
    out[(size_t)(s * 64 + k) * Kpad + c] = (_Float16)v;
  }
}

// ---------------------------------------------------------------------------
// Fused GEMM + bias + BN_SCALE + ReLU via v_wmma_f32_16x16x32_f16.
// Y[M,N] = act((X[M,K] * W[N,K]^T + b) * BN_SCALE). K % 32 == 0, N % 64 == 0.
// Block = 256 threads = 8 waves; block tile 128(M) x 64(N); K chunks of 32
// staged in LDS. Register blocking: each wave keeps one A fragment and 4
// accumulators, issuing 4 back-to-back WMMAs per K-chunk (A reused across 4
// B sub-tiles) -> 4x compute per LDS/barrier round and 4x less X traffic
// from HBM. EXEC stays all-ones through the WMMA region.
// ---------------------------------------------------------------------------
#define GEMM_BM 128
#define GEMM_BN 64
#define GEMM_BK 32
#define GEMM_LS 40  // padded LDS stride in halves

__global__ __launch_bounds__(256) void gemm_bias_act_wmma(
    const _Float16* __restrict__ X, const _Float16* __restrict__ W,
    const float* __restrict__ bias, _Float16* __restrict__ Y,
    int M, int K, int N, int relu) {
  __shared__ _Float16 Xs[GEMM_BM][GEMM_LS];
  __shared__ _Float16 Ws[GEMM_BN][GEMM_LS];

  const int tid = threadIdx.x;
  const int lane = tid & 31;
  const int wave = tid >> 5;
  const int m0 = blockIdx.x * GEMM_BM;
  const int n0 = blockIdx.y * GEMM_BN;

  v8f acc[4] = {{}, {}, {}, {}};

  for (int k0 = 0; k0 < K; k0 += GEMM_BK) {
    // X tile: 128x32 halves; 256 threads x 16 contiguous halves each
    {
      int base = tid * 16;
      int row = base >> 5;          // /32
      int col = base & 31;          // 0 or 16
      const _Float16* src = X + (size_t)(m0 + row) * K + k0 + col;
      if (m0 + row < M) {
        *(v16h*)&Xs[row][col] = *(const v16h*)src;
        if (k0 + GEMM_BK < K) __builtin_prefetch((const void*)(src + GEMM_BK), 0, 1);
      } else {
        v16h z = {};
        *(v16h*)&Xs[row][col] = z;
      }
    }
    // W tile: 64x32 halves; 256 threads x 8 contiguous halves each (b128)
    {
      int base = tid * 8;
      int row = base >> 5;          // 0..63
      int col = base & 31;          // 0,8,16,24
      *(v8h*)&Ws[row][col] = *(const v8h*)(W + (size_t)(n0 + row) * K + k0 + col);
    }
    __syncthreads();

    // A fragment (16x32 f16): lanes 0-15 -> K 0..15, lanes 16-31 -> K 16..31
    v16h a;
    {
      int arow = wave * 16 + (lane & 15);
      int kb = (lane < 16) ? 0 : 16;
#pragma unroll
      for (int i = 0; i < 8; ++i) {
        a[i]     = Xs[arow][kb + i];
        a[i + 8] = Xs[arow][kb + 8 + i];
      }
    }
    // 4 B fragments (32x16 f16 each): lane holds one output column, half K range
    int bcol = lane & 15;
    int kbb = (lane < 16) ? 0 : 16;
#pragma unroll
    for (int nt = 0; nt < 4; ++nt) {
      v16h b;
#pragma unroll
      for (int i = 0; i < 16; ++i) b[i] = Ws[nt * 16 + bcol][kbb + i];
      acc[nt] = __builtin_amdgcn_wmma_f32_16x16x32_f16(
          /*neg_a=*/false, a, /*neg_b=*/false, b,
          /*c_mod=*/(short)0, acc[nt], /*reuse_a=*/false, /*reuse_b=*/false);
    }
    __syncthreads();
  }

  // Epilogue: C/D layout — VGPR r: lanes 0-15 -> M=r, lanes 16-31 -> M=r+8
  int col = lane & 15;
  int mbase = m0 + wave * 16 + ((lane < 16) ? 0 : 8);
#pragma unroll
  for (int nt = 0; nt < 4; ++nt) {
    int n = n0 + nt * 16 + col;
    float bv = bias ? bias[n] : 0.0f;
#pragma unroll
    for (int r = 0; r < 8; ++r) {
      int m = mbase + r;
      if (m < M) {
        float v = (acc[nt][r] + bv) * BN_SCALE_C;
        if (relu) v = fmaxf(v, 0.0f);
        Y[(size_t)m * N + n] = (_Float16)v;
      }
    }
  }
}

// max over nsample=64 grouped rows: [npt*64, C] -> [npt, C]
__global__ void maxpool64_kernel(const _Float16* __restrict__ Yin, _Float16* __restrict__ out,
                                 int npt, int C) {
  int i = blockIdx.x * blockDim.x + threadIdx.x;
  if (i >= npt * C) return;
  int s = i / C, c = i - s * C;
  const _Float16* p = Yin + (size_t)s * 64 * C + c;
  float m = -1e30f;
  for (int k = 0; k < 64; ++k) m = fmaxf(m, (float)p[(size_t)k * C]);
  out[i] = (_Float16)m;
}

// 3-NN of each dense point among sparse points; inverse-distance weights
__global__ void three_nn_kernel(const float* __restrict__ dXyz, int Nd,
                                const float* __restrict__ sXyz, int Ns,
                                int* __restrict__ knnIdx, float* __restrict__ knnW) {
  int n = blockIdx.x * blockDim.x + threadIdx.x;
  if (n >= Nd) return;
  float bx = dXyz[n * 3 + 0], by = dXyz[n * 3 + 1], bz = dXyz[n * 3 + 2];
  float d0 = 1e30f, d1 = 1e30f, d2 = 1e30f;
  int i0 = 0, i1 = 0, i2 = 0;
  for (int j = 0; j < Ns; ++j) {
    float dx = sXyz[j * 3 + 0] - bx;
    float dy = sXyz[j * 3 + 1] - by;
    float dz = sXyz[j * 3 + 2] - bz;
    float dd = dx * dx + dy * dy + dz * dz;
    if (dd < d0)      { d2 = d1; i2 = i1; d1 = d0; i1 = i0; d0 = dd; i0 = j; }
    else if (dd < d1) { d2 = d1; i2 = i1; d1 = dd; i1 = j; }
    else if (dd < d2) { d2 = dd; i2 = j; }
  }
  float w0 = 1.0f / (d0 + 1e-8f), w1 = 1.0f / (d1 + 1e-8f), w2 = 1.0f / (d2 + 1e-8f);
  float s = w0 + w1 + w2;
  knnIdx[n * 3 + 0] = i0; knnIdx[n * 3 + 1] = i1; knnIdx[n * 3 + 2] = i2;
  knnW[n * 3 + 0] = w0 / s; knnW[n * 3 + 1] = w1 / s; knnW[n * 3 + 2] = w2 / s;
}

// build FP input: [interp(Cs) ++ denseFeat(Cd) ++ (xyz?3:0) ++ zeropad] f16 [Nd,Kpad]
__global__ void fp_concat_kernel(const int* __restrict__ knnIdx, const float* __restrict__ knnW,
                                 const _Float16* __restrict__ sFeat, int Cs,
                                 const _Float16* __restrict__ dFeat, int Cd,
                                 const float* __restrict__ dXyz, int addXyz,
                                 _Float16* __restrict__ out, int Nd, int Kpad) {
  int i = blockIdx.x * blockDim.x + threadIdx.x;
  if (i >= Nd * Kpad) return;
  int n = i / Kpad, c = i - n * Kpad;
  float v = 0.0f;
  if (c < Cs) {
    int i0 = knnIdx[n * 3 + 0], i1 = knnIdx[n * 3 + 1], i2 = knnIdx[n * 3 + 2];
    float w0 = knnW[n * 3 + 0], w1 = knnW[n * 3 + 1], w2 = knnW[n * 3 + 2];
    v = w0 * (float)sFeat[(size_t)i0 * Cs + c] +
        w1 * (float)sFeat[(size_t)i1 * Cs + c] +
        w2 * (float)sFeat[(size_t)i2 * Cs + c];
  } else if (c < Cs + Cd) {
    v = (float)dFeat[(size_t)n * Cd + (c - Cs)];
  } else if (addXyz && c < Cs + Cd + 3) {
    v = dXyz[n * 3 + (c - Cs - Cd)];
  }
  out[i] = (_Float16)v;
}

// conf head: sigmoid((x . w + b) * BN_SCALE), X f16 [N,128]
__global__ void conf_head_kernel(const _Float16* __restrict__ X, const float* __restrict__ w,
                                 const float* __restrict__ b, float* __restrict__ conf, int N) {
  int n = blockIdx.x * blockDim.x + threadIdx.x;
  if (n >= N) return;
  float a = 0.0f;
  for (int c = 0; c < 128; ++c) a += (float)X[(size_t)n * 128 + c] * w[c];
  a = (a + b[0]) * BN_SCALE_C;
  conf[n] = 1.0f / (1.0f + expf(-a));
}

// rank-based top-k: rank = #points with larger conf; rank<M gets slot rank
__global__ void topk_select_kernel(const float* __restrict__ conf, int N, int M,
                                   int* __restrict__ selIdx, const float* __restrict__ xyz,
                                   float* __restrict__ outXyz) {
  int p = blockIdx.x * blockDim.x + threadIdx.x;
  if (p >= N) return;
  float v = conf[p];
  int rank = 0;
  for (int q = 0; q < N; ++q) {
    float u = conf[q];
    rank += (u > v) || (u == v && q < p);
  }
  if (rank < M) {
    selIdx[rank] = p;
    outXyz[0 * M + rank] = xyz[p * 3 + 0];
    outXyz[1 * M + rank] = xyz[p * 3 + 1];
    outXyz[2 * M + rank] = xyz[p * 3 + 2];
  }
}

__global__ void gather_rows_kernel(const _Float16* __restrict__ src, const int* __restrict__ idx,
                                   _Float16* __restrict__ dst, int M, int C, int Nmax) {
  int i = blockIdx.x * blockDim.x + threadIdx.x;
  if (i >= M * C) return;
  int r = i / C, c = i - r * C;
  int j = idx[r];
  j = (j < 0) ? 0 : (j >= Nmax ? Nmax - 1 : j);
  dst[i] = src[(size_t)j * C + c];
}

// final linear (no relu): out[o*M + n] = (X[n,:128] . W[o] + b[o]) * BN_SCALE
__global__ void linear_head_kernel(const _Float16* __restrict__ X, const float* __restrict__ W,
                                   const float* __restrict__ b, float* __restrict__ out,
                                   int M, int O) {
  int i = blockIdx.x * blockDim.x + threadIdx.x;
  if (i >= M * O) return;
  int o = i / M, n = i - o * M;
  float a = 0.0f;
  for (int c = 0; c < 128; ++c) a += (float)X[(size_t)n * 128 + c] * W[o * 128 + c];
  out[(size_t)o * M + n] = (a + b[o]) * BN_SCALE_C;
}

// ---------------------------------------------------------------------------
extern "C" void kernel_launch(void* const* d_in, const int* in_sizes, int n_in,
                              void* d_out, int out_size, void* d_ws, size_t ws_size,
                              hipStream_t stream) {
  (void)in_sizes; (void)n_in; (void)out_size; (void)ws_size;
  const int N0 = 12800, NP1 = 5120, NP2 = 1024, NP3 = 256, MSEL = 6400;
  const float* pts = (const float*)d_in[0];

  struct LayerDesc { int wIdx; int cout; int cin; };
  static const LayerDesc L[28] = {
      {1, 128, 6},   {3, 128, 128},  {5, 256, 128},   // SA1
      {7, 256, 259}, {9, 256, 256},  {11, 512, 256},  // SA2
      {13, 512, 515},{15, 512, 512}, {17, 1024, 512}, // SA3
      {19, 1024, 1536}, {21, 1024, 1024},             // FP0
      {23, 512, 1280},  {25, 512, 512},               // FP1
      {27, 256, 518},   {29, 256, 256}, {31, 256, 256}, // FP2
      {33, 512, 256}, {35, 256, 512}, {37, 256, 256}, {39, 128, 256}, // seg
      {41, 512, 256}, {43, 256, 512}, {45, 256, 256}, {47, 128, 256}, // cls
      {49, 512, 256}, {51, 256, 512}, {53, 256, 256}, {55, 128, 256}, // reg
  };
  auto kp = [](int k) { return (k + 31) & ~31; };

  char* wsb = (char*)d_ws;
  size_t off = 0;
  auto alloc = [&](size_t bytes) -> void* {
    void* p = (void*)(wsb + off);
    off += (bytes + 255) & ~(size_t)255;
    return p;
  };

  float*     xyz0    = (float*)alloc((size_t)N0 * 3 * 4);
  _Float16*  feat0   = (_Float16*)alloc((size_t)N0 * 3 * 2);
  float*     minDist = (float*)alloc((size_t)N0 * 4);
  float*     ctr1    = (float*)alloc((size_t)NP1 * 3 * 4);
  float*     ctr2    = (float*)alloc((size_t)NP2 * 3 * 4);
  float*     ctr3    = (float*)alloc((size_t)NP3 * 3 * 4);
  _Float16*  feat1   = (_Float16*)alloc((size_t)NP1 * 256 * 2);
  _Float16*  feat2   = (_Float16*)alloc((size_t)NP2 * 512 * 2);
  _Float16*  feat3   = (_Float16*)alloc((size_t)NP3 * 1024 * 2);
  _Float16*  fpo0    = (_Float16*)alloc((size_t)NP2 * 1024 * 2);
  _Float16*  fpo1    = (_Float16*)alloc((size_t)NP1 * 512 * 2);
  _Float16*  fpo2    = (_Float16*)alloc((size_t)N0 * 256 * 2);
  int*       knnIdx  = (int*)alloc((size_t)N0 * 3 * 4);
  float*     knnW    = (float*)alloc((size_t)N0 * 3 * 4);
  int*       selIdx  = (int*)alloc((size_t)MSEL * 4);
  _Float16*  selFeat = (_Float16*)alloc((size_t)MSEL * 256 * 2);
  _Float16*  G       = (_Float16*)alloc((size_t)65536 * 288 * 2);   // max grouped/concat tile
  _Float16*  A       = (_Float16*)alloc((size_t)327680 * 256 * 2);  // ping
  _Float16*  Bb      = (_Float16*)alloc((size_t)327680 * 128 * 2);  // pong

  // Convert all weights to padded f16 once (deterministic every call).
  _Float16* Wp[28];
  for (int i = 0; i < 28; ++i) {
    int kpad = kp(L[i].cin);
    Wp[i] = (_Float16*)alloc((size_t)L[i].cout * kpad * 2);
    int total = L[i].cout * kpad;
    convert_weights_kernel<<<(total + 255) / 256, 256, 0, stream>>>(
        (const float*)d_in[L[i].wIdx], Wp[i], L[i].cout, L[i].cin, kpad);
  }

  auto gemmL = [&](const _Float16* X, int li, int M, int K, _Float16* Y) {
    int cout = L[li].cout;
    dim3 g((M + GEMM_BM - 1) / GEMM_BM, cout / GEMM_BN);
    gemm_bias_act_wmma<<<g, 256, 0, stream>>>(
        X, Wp[li], (const float*)d_in[L[li].wIdx + 1], Y, M, K, cout, 1);
  };

  split_points_kernel<<<(N0 + 255) / 256, 256, 0, stream>>>(pts, N0, xyz0, feat0);

  // ---- SA1: 12800 -> 5120 centroids, radius 0.02, MLP 6->128->128->256 ----
  fps_kernel<<<1, 1024, 0, stream>>>(xyz0, N0, NP1, ctr1, minDist);
  ball_group_kernel<<<(NP1 + 7) / 8, 256, 0, stream>>>(xyz0, N0, feat0, 3, ctr1, NP1, 0.02f, G, kp(6));
  gemmL(G,  0, NP1 * 64, kp(6), A);
  gemmL(A,  1, NP1 * 64, 128,   Bb);
  gemmL(Bb, 2, NP1 * 64, 128,   A);
  maxpool64_kernel<<<(NP1 * 256 + 255) / 256, 256, 0, stream>>>(A, feat1, NP1, 256);

  // ---- SA2: 5120 -> 1024, radius 0.08, MLP 259->256->256->512 ----
  fps_kernel<<<1, 1024, 0, stream>>>(ctr1, NP1, NP2, ctr2, minDist);
  ball_group_kernel<<<(NP2 + 7) / 8, 256, 0, stream>>>(ctr1, NP1, feat1, 256, ctr2, NP2, 0.08f, G, kp(259));
  gemmL(G,  3, NP2 * 64, kp(259), A);
  gemmL(A,  4, NP2 * 64, 256,     Bb);
  gemmL(Bb, 5, NP2 * 64, 256,     A);
  maxpool64_kernel<<<(NP2 * 512 + 255) / 256, 256, 0, stream>>>(A, feat2, NP2, 512);

  // ---- SA3: 1024 -> 256, radius 0.32, MLP 515->512->512->1024 ----
  fps_kernel<<<1, 1024, 0, stream>>>(ctr2, NP2, NP3, ctr3, minDist);
  ball_group_kernel<<<(NP3 + 7) / 8, 256, 0, stream>>>(ctr2, NP2, feat2, 512, ctr3, NP3, 0.32f, G, kp(515));
  gemmL(G,  6, NP3 * 64, kp(515), A);
  gemmL(A,  7, NP3 * 64, 512,     Bb);
  gemmL(Bb, 8, NP3 * 64, 512,     A);
  maxpool64_kernel<<<(NP3 * 1024 + 255) / 256, 256, 0, stream>>>(A, feat3, NP3, 1024);

  // ---- FP0: dense=level2 (1024 pts), sparse=level3; 1536 -> 1024 -> 1024 ----
  three_nn_kernel<<<(NP2 + 255) / 256, 256, 0, stream>>>(ctr2, NP2, ctr3, NP3, knnIdx, knnW);
  fp_concat_kernel<<<(NP2 * 1536 + 255) / 256, 256, 0, stream>>>(
      knnIdx, knnW, feat3, 1024, feat2, 512, ctr2, 0, G, NP2, 1536);
  gemmL(G,  9, NP2, 1536, A);
  gemmL(A, 10, NP2, 1024, fpo0);

  // ---- FP1: dense=level1 (5120 pts); 1280 -> 512 -> 512 ----
  three_nn_kernel<<<(NP1 + 255) / 256, 256, 0, stream>>>(ctr1, NP1, ctr2, NP2, knnIdx, knnW);
  fp_concat_kernel<<<(NP1 * 1280 + 255) / 256, 256, 0, stream>>>(
      knnIdx, knnW, fpo0, 1024, feat1, 256, ctr1, 0, G, NP1, 1280);
  gemmL(G, 11, NP1, 1280, A);
  gemmL(A, 12, NP1, 512,  fpo1);

  // ---- FP2: dense=level0 (12800 pts); 518(pad 544) -> 256 -> 256 -> 256 ----
  three_nn_kernel<<<(N0 + 255) / 256, 256, 0, stream>>>(xyz0, N0, ctr1, NP1, knnIdx, knnW);
  fp_concat_kernel<<<(N0 * kp(518) + 255) / 256, 256, 0, stream>>>(
      knnIdx, knnW, fpo1, 512, feat0, 3, xyz0, 1, G, N0, kp(518));
  gemmL(G,  13, N0, kp(518), A);
  gemmL(A,  14, N0, 256,     Bb);
  gemmL(Bb, 15, N0, 256,     fpo2);

  // ---- seg head: 256 -> 512 -> 256 -> 256 -> 128, then conf + top-k ----
  gemmL(fpo2, 16, N0, 256, A);
  gemmL(A,    17, N0, 512, Bb);
  gemmL(Bb,   18, N0, 256, A);
  gemmL(A,    19, N0, 256, Bb);
  conf_head_kernel<<<(N0 + 255) / 256, 256, 0, stream>>>(
      Bb, (const float*)d_in[57], (const float*)d_in[58], (float*)d_out, N0);
  topk_select_kernel<<<(N0 + 255) / 256, 256, 0, stream>>>(
      (const float*)d_out, N0, MSEL, selIdx, xyz0,
      (float*)d_out + 12800 + 51200 + 230400);
  gather_rows_kernel<<<(MSEL * 256 + 255) / 256, 256, 0, stream>>>(
      fpo2, selIdx, selFeat, MSEL, 256, N0);

  // ---- cls head on selected 6400 points ----
  gemmL(selFeat, 20, MSEL, 256, A);
  gemmL(A,       21, MSEL, 512, Bb);
  gemmL(Bb,      22, MSEL, 256, A);
  gemmL(A,       23, MSEL, 256, Bb);
  linear_head_kernel<<<(MSEL * 8 + 255) / 256, 256, 0, stream>>>(
      Bb, (const float*)d_in[59], (const float*)d_in[60], (float*)d_out + 12800, MSEL, 8);

  // ---- reg head ----
  gemmL(selFeat, 24, MSEL, 256, A);
  gemmL(A,       25, MSEL, 512, Bb);
  gemmL(Bb,      26, MSEL, 256, A);
  gemmL(A,       27, MSEL, 256, Bb);
  linear_head_kernel<<<(MSEL * 36 + 255) / 256, 256, 0, stream>>>(
      Bb, (const float*)d_in[61], (const float*)d_in[62],
      (float*)d_out + 12800 + 51200, MSEL, 36);
}
